// CellBox_69071664054304
// MI455X (gfx1250) — matrix operand order repaired and checked by reading, not compiled, not run on hardware
//
#include <hip/hip_runtime.h>
#include <math.h>

// ---------------------------------------------------------------------------
// CellBox Heun integrator on MI455X (gfx1250, wave32, WMMA).
// 81 GEMMs of (1024 x 2048) = (1024 x 2048[bf16]) @ (2048 x 2048[bf16])^T
// with f32 accumulate, fused tanh/Heun epilogues. W masked+bf16-converted once.
// Global->LDS staging uses CDNA5 async loads (ASYNCcnt) instead of VGPR relay.
// ---------------------------------------------------------------------------

typedef __attribute__((ext_vector_type(16))) __bf16        v16bf;
typedef __attribute__((ext_vector_type(8)))  float         v8f;
typedef __attribute__((ext_vector_type(4)))  unsigned int  v4u;

// gcc-vector int4 matching the async-builtin's parameter type
typedef int v4i_t __attribute__((vector_size(16)));
typedef __attribute__((address_space(1))) v4i_t* g_v4i_p;   // global int4*
typedef __attribute__((address_space(3))) v4i_t* l_v4i_p;   // LDS    int4*

#define NX 2048   // neurons (GEMM N and K)
#define NB 1024   // batch   (GEMM M)
#define N_PROT 1600
#define N_ACT  1800
#define DTC 0.1f

union Frag { v16bf bf; v4u u[2]; };

__device__ __forceinline__ float softplusf(float x) {
    return (x > 15.f) ? x : log1pf(expf(x));
}

// ---- CDNA5 async global->LDS copy (16B per lane, ASYNCcnt-tracked) --------
__device__ __forceinline__ void async_cp16(const void* g, void* l) {
#if __has_builtin(__builtin_amdgcn_global_load_async_to_lds_b128)
    __builtin_amdgcn_global_load_async_to_lds_b128(
        (g_v4i_p)(void*)g, (l_v4i_p)l, 0, 0);
#else
    unsigned lofs = (unsigned)(size_t)(__attribute__((address_space(3))) char*)l;
    asm volatile("global_load_async_to_lds_b128 %0, %1, off"
                 :: "v"(lofs), "v"(g) : "memory");
#endif
}

__device__ __forceinline__ void wait_async0() {
#if __has_builtin(__builtin_amdgcn_s_wait_asynccnt)
    __builtin_amdgcn_s_wait_asynccnt(0);
#else
    asm volatile("s_wait_asynccnt 0x0" ::: "memory");
#endif
}

// ---------------------------------------------------------------------------
// Prep kernels (one-time)
// ---------------------------------------------------------------------------
__global__ void prep_params(const float* __restrict__ ar, const float* __restrict__ er,
                            float* __restrict__ alpha, float* __restrict__ eps) {
    int i = blockIdx.x * 256 + threadIdx.x;
    if (i < NX) { alpha[i] = softplusf(ar[i]); eps[i] = softplusf(er[i]); }
}

__global__ void prep_W(const float* __restrict__ W, __bf16* __restrict__ Wb) {
    int i = blockIdx.x * 256 + threadIdx.x;   // over NX*NX
    int m = i >> 11, k = i & (NX - 1);
    bool z = (m == k) || (m >= N_ACT) || (k >= N_PROT && k < N_ACT) ||
             (m >= N_PROT && m < N_ACT && k >= N_ACT);
    Wb[i] = z ? (__bf16)0.f : (__bf16)W[i];
}

__global__ void prep_x0(const float* __restrict__ y0,
                        float* __restrict__ xT, __bf16* __restrict__ xb) {
    int i = blockIdx.x * 256 + threadIdx.x;   // over NB*NX, (n,m) layout
    int n = i >> 11, m = i & (NX - 1);
    float v = y0[(size_t)m * NB + n];         // y0 is (NX, NB)
    xT[i] = v; xb[i] = (__bf16)v;
}

// ---------------------------------------------------------------------------
// Fused GEMM + dxdt/Heun epilogue.
//   phase 0: p0=x      -> o0=d1, o1=xtmp(f32), ob=xtmp(bf16)
//   phase 1: p0=xtmp, p1=x, p2=d1 -> o0=x_new(f32), ob=x_new(bf16)
//   phase 2: p0=x40, p1=x39 -> out = [mean; sd; dx] (6144x1024) ++ yhat (1024x2048)
// ---------------------------------------------------------------------------
__global__ __launch_bounds__(256) void cellbox_gemm(
    const __bf16* __restrict__ Abf,   // (NB, NX) bf16: current GEMM input state
    const __bf16* __restrict__ Bbf,   // (NX, NX) bf16: masked W, row = out neuron
    const float*  __restrict__ mu,    // (NB, NX)
    const float*  __restrict__ alpha, // (NX)
    const float*  __restrict__ eps,   // (NX)
    const float*  __restrict__ p0, const float* __restrict__ p1,
    const float*  __restrict__ p2,
    float* __restrict__ o0, float* __restrict__ o1,
    __bf16* __restrict__ ob, float* __restrict__ out, int phase)
{
    // [buf][A/B][row][five 16B chunks]; 80B row stride => conflict-free b128
    __shared__ v4u smem[2][2][128][5];

    const int tid  = threadIdx.x;
    const int bm   = blockIdx.x & 7;     // batch-block   (NB/128 = 8)
    const int bn   = blockIdx.x >> 3;    // neuron-block  (NX/128 = 16)
    const int lane = tid & 31, w = tid >> 5;
    const int wm   = w >> 2;             // 0..1 : 64-row slice (batch)
    const int wn   = w & 3;              // 0..3 : 32-col slice (neuron)
    const int h    = lane >> 4, l16 = lane & 15;

    const int rowL = tid >> 2;           // 0..63 loader row
    const int c4   = tid & 3;            // 16B chunk within 64B K-slab

    const size_t arow0 = (size_t)(bm * 128 + rowL) * NX + c4 * 8;
    const size_t arow1 = arow0 + (size_t)64 * NX;
    const size_t brow0 = (size_t)(bn * 128 + rowL) * NX + c4 * 8;
    const size_t brow1 = brow0 + (size_t)64 * NX;

    v8f acc[4][2] = {};

    // async-stage one 32-deep K-block into LDS buffer `buf`
    auto issue = [&](int k0, int buf) {
        async_cp16(Abf + arow0 + k0, &smem[buf][0][rowL][c4]);
        async_cp16(Abf + arow1 + k0, &smem[buf][0][rowL + 64][c4]);
        async_cp16(Bbf + brow0 + k0, &smem[buf][1][rowL][c4]);
        async_cp16(Bbf + brow1 + k0, &smem[buf][1][rowL + 64][c4]);
    };

    // consume LDS buffer `cur`: 6 fragment loads, 8 WMMAs
    auto compute = [&](int cur) {
        Frag a[4], b[2];
        #pragma unroll
        for (int mt = 0; mt < 4; ++mt) {          // A 16x32: chunks h and 2+h
            const int r = wm * 64 + mt * 16 + l16;
            a[mt].u[0] = smem[cur][0][r][h];
            a[mt].u[1] = smem[cur][0][r][2 + h];
        }
        #pragma unroll
        for (int nt = 0; nt < 2; ++nt) {          // B 32x16: chunks h and 2+h
            const int r = wn * 32 + nt * 16 + l16;
            b[nt].u[0] = smem[cur][1][r][h];
            b[nt].u[1] = smem[cur][1][r][2 + h];
        }
        #pragma unroll
        for (int mt = 0; mt < 4; ++mt)
            #pragma unroll
            for (int nt = 0; nt < 2; ++nt)
                acc[mt][nt] = __builtin_amdgcn_wmma_f32_16x16x32_bf16(
                    false, a[mt].bf, false, b[nt].bf,
                    (short)0, acc[mt][nt], false, false);
    };

    const int K_ITERS = NX / 32;   // 64

    issue(0, 0);
    wait_async0();
    __syncthreads();

    for (int kk = 0; kk < K_ITERS - 1; ++kk) {
        const int cur = kk & 1;
        issue((kk + 1) * 32, cur ^ 1);   // DMA next block while computing
        compute(cur);
        wait_async0();                   // own async loads done
        __syncthreads();                 // everyone's loads visible
    }
    compute((K_ITERS - 1) & 1);

    // Epilogue: lane holds D[n][m] with n = base+v+8h (per VGPR), m = base+l16.
    #pragma unroll
    for (int mt = 0; mt < 4; ++mt) {
        const int n_base = bm * 128 + wm * 64 + mt * 16 + 8 * h;
        #pragma unroll
        for (int nt = 0; nt < 2; ++nt) {
            const int m_i = bn * 128 + wn * 32 + nt * 16 + l16;
            const float e  = eps[m_i];
            const float al = alpha[m_i];
            #pragma unroll
            for (int v = 0; v < 8; ++v) {
                const int    n   = n_base + v;
                const size_t idx = (size_t)n * NX + m_i;
                const float  t   = tanhf(acc[mt][nt][v] + mu[idx]);
                if (phase == 0) {
                    const float x  = p0[idx];
                    const float d1 = e * t - al * x;
                    o0[idx] = d1;
                    const float xt = x + DTC * d1;
                    o1[idx] = xt;
                    ob[idx] = (__bf16)xt;
                } else if (phase == 1) {
                    const float xt = p0[idx];
                    const float d2 = e * t - al * xt;
                    const float xn = p1[idx] + 0.5f * DTC * (p2[idx] + d2);
                    o0[idx] = xn;
                    ob[idx] = (__bf16)xn;
                } else {
                    const float x40 = p0[idx], x39 = p1[idx];
                    const float dxv = e * t - al * x40;
                    const size_t cm = (size_t)m_i * NB + n;
                    out[cm]                            = 0.5f * (x39 + x40);
                    out[(size_t)NX * NB + cm]          = fabsf(x40 - x39) * 0.70710678118654752f;
                    out[2 * (size_t)NX * NB + cm]      = dxv;
                    out[3 * (size_t)NX * NB + (size_t)n * NX + m_i] = x40;
                }
            }
        }
    }
}

// ---------------------------------------------------------------------------
// Host launcher: 3 prep + 80 Heun-half GEMMs + 1 final GEMM, all on `stream`.
// Workspace layout (52 MiB + 16 KiB):
//   Wb 8M | xb0 4M | xb1 4M | xtmpb 4M | xT0 8M | xT1 8M | xtmpT 8M | d1T 8M
//   | alpha 8K | eps 8K
// ---------------------------------------------------------------------------
extern "C" void kernel_launch(void* const* d_in, const int* in_sizes, int n_in,
                              void* d_out, int out_size, void* d_ws, size_t ws_size,
                              hipStream_t stream) {
    (void)in_sizes; (void)n_in; (void)out_size; (void)ws_size;
    const float* y0 = (const float*)d_in[0];
    const float* mu = (const float*)d_in[1];
    const float* W  = (const float*)d_in[2];
    const float* ar = (const float*)d_in[3];
    const float* er = (const float*)d_in[4];
    float* out = (float*)d_out;

    char* p = (char*)d_ws;
    __bf16* Wb    = (__bf16*)(p);
    __bf16* xb0   = (__bf16*)(p + ( 8ull << 20));
    __bf16* xb1   = (__bf16*)(p + (12ull << 20));
    __bf16* xtb   = (__bf16*)(p + (16ull << 20));
    float*  xT0   = (float*) (p + (20ull << 20));
    float*  xT1   = (float*) (p + (28ull << 20));
    float*  xtT   = (float*) (p + (36ull << 20));
    float*  d1T   = (float*) (p + (44ull << 20));
    float*  alpha = (float*) (p + (52ull << 20));
    float*  eps   = (float*) (p + (52ull << 20) + 8192);

    prep_params<<<8, 256, 0, stream>>>(ar, er, alpha, eps);
    prep_W<<<(NX * NX) / 256, 256, 0, stream>>>(W, Wb);
    prep_x0<<<(NB * NX) / 256, 256, 0, stream>>>(y0, xT0, xb0);

    __bf16* xb[2] = {xb0, xb1};
    float*  xT[2] = {xT0, xT1};

    for (int s = 1; s <= 40; ++s) {
        const int cur = (s - 1) & 1, nxt = s & 1;
        // half-step A: d1 = dxdt(x); xtmp = x + DT*d1
        cellbox_gemm<<<128, 256, 0, stream>>>(xb[cur], Wb, mu, alpha, eps,
            xT[cur], nullptr, nullptr, d1T, xtT, xtb, nullptr, 0);
        // half-step B: d2 = dxdt(xtmp); x = x + DT/2*(d1+d2)
        cellbox_gemm<<<128, 256, 0, stream>>>(xtb, Wb, mu, alpha, eps,
            xtT, xT[cur], d1T, xT[nxt], nullptr, xb[nxt], nullptr, 1);
    }
    // final: x40 in buf0, x39 in buf1 -> metrics + yhat
    cellbox_gemm<<<128, 256, 0, stream>>>(xb[0], Wb, mu, alpha, eps,
        xT[0], xT[1], nullptr, nullptr, nullptr, nullptr, out, 2);
}
